// RotaryMultiHeadAttention_76639396430329
// MI455X (gfx1250) — matrix operand order repaired and checked by reading, not compiled, not run on hardware
//
#include <hip/hip_runtime.h>
#include <hip/hip_bf16.h>
#include <math.h>

typedef __attribute__((ext_vector_type(16))) __bf16 v16bf;
typedef __attribute__((ext_vector_type(8)))  float  v8f;

#define D_MODEL  2048
#define N_HEADS  16
#define HEAD_DIM 128
#define SEQ      2048
#define BATCH    2
#define M_TOT    (BATCH*SEQ)   // 4096

// ---- gfx1250 async global->LDS path (guarded; falls back to uint4 copy) ----
#if defined(__has_builtin)
#  if __has_builtin(__builtin_amdgcn_global_load_async_to_lds_b128)
#    define ATHENA_ASYNC_LDS 1
#  endif
#endif
#ifndef ATHENA_ASYNC_LDS
#  define ATHENA_ASYNC_LDS 0
#endif

typedef int athena_v4i __attribute__((vector_size(16)));
typedef __attribute__((address_space(1))) athena_v4i* athena_gv4i_p;
typedef __attribute__((address_space(3))) athena_v4i* athena_lv4i_p;

static __device__ __forceinline__ void g2l_b128(const __bf16* __restrict__ g,
                                                __bf16* l) {
#if ATHENA_ASYNC_LDS
  __builtin_amdgcn_global_load_async_to_lds_b128(
      (athena_gv4i_p)g, (athena_lv4i_p)l, 0, 0);
#else
  *reinterpret_cast<uint4*>(l) = *reinterpret_cast<const uint4*>(g);
#endif
}

static __device__ __forceinline__ void wait_async_lds() {
#if ATHENA_ASYNC_LDS
#  if __has_builtin(__builtin_amdgcn_s_wait_asynccnt)
  __builtin_amdgcn_s_wait_asynccnt(0);
#  else
  asm volatile("s_wait_asynccnt 0" ::: "memory");
#  endif
#endif
}

static __device__ __forceinline__ v8f zero8() {
  v8f z;
  #pragma unroll
  for (int i = 0; i < 8; ++i) z[i] = 0.0f;
  return z;
}

// ---------------------------------------------------------------------------
// C[M,N] = X[M,K] * W[N,K]^T + bias, via v_wmma_f32_16x16x32_bf16.
// Block: 128 threads = 4 waves. Tile 64(M) x 64(N); wave w owns rows [w*16,+16).
// OUT_MODE 0: f32 row-major [M, D_MODEL]  (Q/K pre-RoPE, final output)
// OUT_MODE 1: bf16 [B, H, S, Dh]          (V, attention-ready layout)
// ---------------------------------------------------------------------------
template<typename TIn, int OUT_MODE>
__global__ __launch_bounds__(128) void gemm_xwt_kernel(
    const TIn* __restrict__ X, const float* __restrict__ W,
    const float* __restrict__ bias, void* __restrict__ outp)
{
  __shared__ __bf16 lds_a[64][32];   // [m][k] row-major
  __shared__ __bf16 lds_b[32][64];   // [k][n] == W tile transposed

  const int tid  = threadIdx.x;
  const int wave = tid >> 5;
  const int lane = tid & 31;
  const int m0 = blockIdx.y * 64;
  const int n0 = blockIdx.x * 64;

  v8f acc[4];
  #pragma unroll
  for (int t = 0; t < 4; ++t) acc[t] = zero8();

  const int aRow = (tid * 16) >> 5;   // 0..63 (half-row of 16 k per thread)
  const int aK   = (tid * 16) & 31;   // 0 or 16
  const int mrow   = lane & 15;
  const int kchunk = (lane >> 4) * 8;

  for (int kb = 0; kb < D_MODEL; kb += 32) {
    // Stage A tile (bf16), contiguous global reads + b128 LDS stores.
    const TIn* xs = X + (size_t)(m0 + aRow) * D_MODEL + kb + aK;
    if (sizeof(TIn) == 2) {
      // already bf16: async DMA straight into LDS (gfx1250 path)
      g2l_b128((const __bf16*)(const void*)xs,     &lds_a[aRow][aK]);
      g2l_b128((const __bf16*)(const void*)xs + 8, &lds_a[aRow][aK + 8]);
    } else {
      __bf16 tmp[16];
      #pragma unroll
      for (int j = 0; j < 16; ++j) tmp[j] = (__bf16)xs[j];
      *reinterpret_cast<uint4*>(&lds_a[aRow][aK])     = *reinterpret_cast<uint4*>(&tmp[0]);
      *reinterpret_cast<uint4*>(&lds_a[aRow][aK + 8]) = *reinterpret_cast<uint4*>(&tmp[8]);
    }
    // Stage W tile transposed: lds_b[k][n] = W[n0+n][kb+k].
    const float* wsrc = W + (size_t)(n0 + aRow) * D_MODEL + kb + aK;
    #pragma unroll
    for (int j = 0; j < 16; ++j) lds_b[aK + j][aRow] = (__bf16)wsrc[j];

    if (kb + 32 < D_MODEL) {                 // gfx1250 global_prefetch_b8
      __builtin_prefetch(xs + 32, 0, 0);
      __builtin_prefetch(wsrc + 32, 0, 0);
    }
    wait_async_lds();
    __syncthreads();

    // A fragment: lane holds row m, K chunks [k0, k0+8) and [k0+16, k0+24).
    v16bf a;
    #pragma unroll
    for (int j = 0; j < 8; ++j) {
      a[j]     = lds_a[wave*16 + mrow][kchunk + j];
      a[8 + j] = lds_a[wave*16 + mrow][kchunk + 16 + j];
    }
    #pragma unroll
    for (int t = 0; t < 4; ++t) {
      // B fragment: lane = N column, 16 K values in table order.
      v16bf bfr;
      #pragma unroll
      for (int j = 0; j < 16; ++j) bfr[j] = lds_b[j < 8 ? kchunk + j : kchunk + 8 + j][t*16 + mrow];
      acc[t] = __builtin_amdgcn_wmma_f32_16x16x32_bf16(
                   false, a, false, bfr, (short)0, acc[t], false, false);
    }
    __syncthreads();
  }

  // Epilogue. C/D layout: VGPR r, lane l -> M = r + (l>>4)*8, N = l&15.
  #pragma unroll
  for (int t = 0; t < 4; ++t) {
    #pragma unroll
    for (int r = 0; r < 8; ++r) {
      const int mg = m0 + wave*16 + r + ((lane >> 4) << 3);
      const int ng = n0 + t*16 + (lane & 15);
      const float v = acc[t][r] + bias[ng];
      if (OUT_MODE == 0) {
        ((float*)outp)[(size_t)mg * D_MODEL + ng] = v;
      } else {
        const int bb = mg >> 11, ss = mg & 2047;
        const int hh = ng >> 7,  dd = ng & 127;
        ((__bf16*)outp)[((((size_t)bb*N_HEADS + hh)*SEQ + ss) << 7) + dd] = (__bf16)v;
      }
    }
  }
}

// ---------------------------------------------------------------------------
// RoPE + permute [B*S, D] f32 -> [B, H, S, Dh] bf16.
// out[d] = x[d]*cos + rh[d]*sin, rh[d] = d<64 ? -x[2d+1] : x[2(d-64)].
// ---------------------------------------------------------------------------
__global__ __launch_bounds__(256) void rope_permute_kernel(
    const float* __restrict__ src, __bf16* __restrict__ dst)
{
  const int idx = blockIdx.x * 256 + threadIdx.x;   // < B*S*D
  const int f  = idx & (D_MODEL - 1);
  const int bs = idx >> 11;
  const int d  = f & (HEAD_DIM - 1);
  const int h  = f >> 7;
  const int s  = bs & (SEQ - 1);
  const int b  = bs >> 11;
  const float* row = src + (size_t)bs * D_MODEL + h * HEAD_DIM;
  const float x  = row[d];
  const float rh = (d < 64) ? -row[2*d + 1] : row[2*(d - 64)];
  // inv_freq[d] = 10000^(-d/1024) = exp(-d * ln(10000)/1024)
  const float ang = (float)s * __expf(-(float)d * 0.008994473019508355f);
  float sv, cv; __sincosf(ang, &sv, &cv);
  dst[(((size_t)b * N_HEADS + h) * SEQ + s) * HEAD_DIM + d] = (__bf16)(x*cv + rh*sv);
}

// ---------------------------------------------------------------------------
// Flash attention. Block = 256 threads (8 waves), one (b, h, 128-q-row) tile.
// Wave owns 16 q rows; kv tiles of 64; scores/ctx accumulated in WMMA C layout.
// V tile staged via async global->LDS DMA; K tile staged transposed for B-frags.
// ---------------------------------------------------------------------------
__global__ __launch_bounds__(256) void flash_attn_kernel(
    const __bf16* __restrict__ Qb, const __bf16* __restrict__ Kb,
    const __bf16* __restrict__ Vb, __bf16* __restrict__ Cb)
{
  __shared__ __bf16 lds_kt[HEAD_DIM][64];   // K^T tile: [dh][kv]
  __shared__ __bf16 lds_v[64][HEAD_DIM];    // V tile:   [kv][dh]
  __shared__ __bf16 lds_p[8][16][64];       // per-wave P staging (C->A relayout)

  const int tid  = threadIdx.x;
  const int wave = tid >> 5;
  const int lane = tid & 31;
  const int b = blockIdx.z;
  const int h = blockIdx.y;
  const int q0 = blockIdx.x * 128 + wave * 16;

  const size_t headBase = ((size_t)b * N_HEADS + h) * SEQ;
  const float scale = 0.08838834764831845f;   // 1/sqrt(128)

  const int mrow   = lane & 15;
  const int kchunk = (lane >> 4) * 8;

  // Q fragments: 4 chunks of 32 along Dh, kept in registers all kernel.
  v16bf qf[4];
  const __bf16* qptr = Qb + (headBase + q0 + mrow) * HEAD_DIM;
  #pragma unroll
  for (int f = 0; f < 4; ++f) {
    #pragma unroll
    for (int j = 0; j < 8; ++j) {
      qf[f][j]     = qptr[f*32 + kchunk + j];
      qf[f][8 + j] = qptr[f*32 + kchunk + 16 + j];
    }
  }

  v8f ctx[8];
  #pragma unroll
  for (int t = 0; t < 8; ++t) ctx[t] = zero8();
  float mprev[8], lsum[8];
  #pragma unroll
  for (int r = 0; r < 8; ++r) { mprev[r] = -1e30f; lsum[r] = 0.0f; }

  for (int kv0 = 0; kv0 < SEQ; kv0 += 64) {
    const __bf16* kbase = Kb + (headBase + kv0) * HEAD_DIM;
    const __bf16* vbase = Vb + (headBase + kv0) * HEAD_DIM;

    // V tile: straight row-major copy -> async DMA, 4 x b128 per thread.
    {
      const int i0 = tid * 32;               // 32 bf16 = 64B per thread
      #pragma unroll
      for (int c = 0; c < 4; ++c)
        g2l_b128(vbase + i0 + c*8, &lds_v[0][0] + i0 + c*8);
    }
    // K tile: transposed store [dh][kv]; each lane reads its kv column for B-frags.
    {
      const int e0 = tid * 32;
      #pragma unroll
      for (int j = 0; j < 32; ++j) {
        const int i = e0 + j;                // 0..8191
        const int kv = i >> 7, dh = i & 127;
        lds_kt[dh][kv] = kbase[i];
      }
    }
    wait_async_lds();
    __syncthreads();

    // S = Q K^T : contraction over Dh in 4 chunks of 32.
    v8f sacc[4];
    #pragma unroll
    for (int t = 0; t < 4; ++t) sacc[t] = zero8();
    #pragma unroll
    for (int f = 0; f < 4; ++f) {
      #pragma unroll
      for (int t = 0; t < 4; ++t) {
        v16bf bfr;   // lane = kv column of the 32x16 B sub-tile
        #pragma unroll
        for (int j = 0; j < 16; ++j) bfr[j] = lds_kt[f*32 + (j < 8 ? kchunk + j : kchunk + 8 + j)][t*16 + mrow];
        sacc[t] = __builtin_amdgcn_wmma_f32_16x16x32_bf16(
                      false, qf[f], false, bfr, (short)0, sacc[t], false, false);
      }
    }

    // Online softmax per accumulator row (lanes 0..15 of each half hold N).
    #pragma unroll
    for (int r = 0; r < 8; ++r) {
      float lm = -1e30f;
      #pragma unroll
      for (int t = 0; t < 4; ++t) lm = fmaxf(lm, sacc[t][r] * scale);
      lm = fmaxf(lm, __shfl_xor(lm, 1, 32));
      lm = fmaxf(lm, __shfl_xor(lm, 2, 32));
      lm = fmaxf(lm, __shfl_xor(lm, 4, 32));
      lm = fmaxf(lm, __shfl_xor(lm, 8, 32));
      const float mnew  = fmaxf(mprev[r], lm);
      const float alpha = __expf(mprev[r] - mnew);
      float ps = 0.0f;
      const int prow = r + ((lane >> 4) << 3);
      #pragma unroll
      for (int t = 0; t < 4; ++t) {
        const float p = __expf(sacc[t][r] * scale - mnew);
        lds_p[wave][prow][t*16 + (lane & 15)] = (__bf16)p;
        ps += p;
      }
      ps += __shfl_xor(ps, 1, 32);
      ps += __shfl_xor(ps, 2, 32);
      ps += __shfl_xor(ps, 4, 32);
      ps += __shfl_xor(ps, 8, 32);
      lsum[r]  = lsum[r] * alpha + ps;
      mprev[r] = mnew;
      #pragma unroll
      for (int t = 0; t < 8; ++t) ctx[t][r] *= alpha;
    }

    // ctx += P V : contraction over kv = 64 in 2 chunks of 32.
    #pragma unroll
    for (int step = 0; step < 2; ++step) {
      v16bf pf;
      #pragma unroll
      for (int j = 0; j < 8; ++j) {
        pf[j]     = lds_p[wave][mrow][step*32 + kchunk + j];
        pf[8 + j] = lds_p[wave][mrow][step*32 + kchunk + 16 + j];
      }
      #pragma unroll
      for (int t = 0; t < 8; ++t) {
        v16bf bfr;   // lane = dh column of the 32x16 B sub-tile
        #pragma unroll
        for (int j = 0; j < 16; ++j) bfr[j] = lds_v[step*32 + (j < 8 ? kchunk + j : kchunk + 8 + j)][t*16 + mrow];
        ctx[t] = __builtin_amdgcn_wmma_f32_16x16x32_bf16(
                     false, pf, false, bfr, (short)0, ctx[t], false, false);
      }
    }
    __syncthreads();
  }

  // Normalize and write context as bf16 [B, S, D] (row = b*S+q, col = h*128+d).
  #pragma unroll
  for (int t = 0; t < 8; ++t) {
    #pragma unroll
    for (int r = 0; r < 8; ++r) {
      const int qg = q0 + r + ((lane >> 4) << 3);
      const int d  = t*16 + (lane & 15);
      const float o = ctx[t][r] / lsum[r];
      Cb[((size_t)b * SEQ + qg) * D_MODEL + h * HEAD_DIM + d] = (__bf16)o;
    }
  }
}

// ---------------------------------------------------------------------------
extern "C" void kernel_launch(void* const* d_in, const int* in_sizes, int n_in,
                              void* d_out, int out_size, void* d_ws, size_t ws_size,
                              hipStream_t stream) {
  const float* query = (const float*)d_in[0];
  const float* key_  = (const float*)d_in[1];
  const float* value = (const float*)d_in[2];
  const float* Wq = (const float*)d_in[3];
  const float* bq = (const float*)d_in[4];
  const float* Wk = (const float*)d_in[5];
  const float* bk = (const float*)d_in[6];
  const float* Wv = (const float*)d_in[7];
  const float* bv = (const float*)d_in[8];
  const float* Wo = (const float*)d_in[9];
  const float* bo = (const float*)d_in[10];

  char* ws = (char*)d_ws;
  float*  qf32 = (float*) (ws);                 // 32 MB
  float*  kf32 = (float*) (ws +  33554432);     // 32 MB
  __bf16* qbf  = (__bf16*)(ws +  67108864);     // 16 MB  [B,H,S,Dh]
  __bf16* kbf  = (__bf16*)(ws +  83886080);     // 16 MB
  __bf16* vbf  = (__bf16*)(ws + 100663296);     // 16 MB
  __bf16* cbf  = (__bf16*)(ws + 117440512);     // 16 MB  bf16 [B*S, D]

  dim3 ggrid(D_MODEL / 64, M_TOT / 64);         // (32, 64)

  gemm_xwt_kernel<float, 0><<<ggrid, 128, 0, stream>>>(query, Wq, bq, qf32);
  gemm_xwt_kernel<float, 0><<<ggrid, 128, 0, stream>>>(key_,  Wk, bk, kf32);
  gemm_xwt_kernel<float, 1><<<ggrid, 128, 0, stream>>>(value, Wv, bv, vbf);

  const int total = M_TOT * D_MODEL;
  rope_permute_kernel<<<total / 256, 256, 0, stream>>>(qf32, qbf);
  rope_permute_kernel<<<total / 256, 256, 0, stream>>>(kf32, kbf);

  flash_attn_kernel<<<dim3(SEQ / 128, N_HEADS, BATCH), 256, 0, stream>>>(
      qbf, kbf, vbf, cbf);

  gemm_xwt_kernel<__bf16, 0><<<ggrid, 128, 0, stream>>>(cbf, Wo, bo, (float*)d_out);
}
